// Gnn_54348516163768
// MI455X (gfx1250) — matrix operand order repaired and checked by reading, not compiled, not run on hardware
//
#include <hip/hip_runtime.h>

// ---------------------------------------------------------------------------
// GNN forward for MI455X (gfx1250, wave32, WMMA).
//  Phase 1: 7 FC layers as f16 WMMA GEMMs with fused BatchNorm-stat
//           accumulation in the epilogue and BN+LeakyReLU folded into the
//           next layer's A-tile load (bias cancels under training-mode BN).
//  Phase 2: 4 SAGEConv layers via edge-scatter fp32 atomics + per-node VALU
//           (8x8 matmuls are far below the WMMA payoff threshold).
// ---------------------------------------------------------------------------

typedef __attribute__((ext_vector_type(16))) _Float16 v16h;
typedef __attribute__((ext_vector_type(8)))  _Float16 v8h;
typedef __attribute__((ext_vector_type(8)))  float    v8f;

#define EPSBN   1e-5f
#define SLOPE   0.1f
#define LDSS    48            // LDS row stride in halves (96B: 16B-aligned, bank-spread)

__device__ __forceinline__ float lrelu(float f) { return f >= 0.f ? f : SLOPE * f; }

// global fp32 atomic add without CAS loop (global_atomic_add_f32)
__device__ __forceinline__ void gAtomAdd(float* p, float v) { unsafeAtomicAdd(p, v); }

// ---------------------------------------------------------------------------
__global__ void zero_f32(float* __restrict__ p, int count) {
    int i = blockIdx.x * blockDim.x + threadIdx.x;
    if (i < count) p[i] = 0.f;
}

// ---------------------------------------------------------------------------
// FC GEMM: Y[m, j] = sum_k A[m, k] * W[k, j]   (bias omitted: cancels in BN)
//   MODE 0: A built on the fly from lf / is_subgoal / is_target (fp32)
//   MODE 1: A = lrelu(Yprev * prevScale[k] + prevShift[k]) from f16 buffer
// Block: 256 threads = 8 waves, wave grid 2(M) x 4(N), each wave 32x32 via
// 2x2 v_wmma_f32_16x16x32_f16.  Block tile: BM=64, BN=128, BK=32, double-buf.
// Epilogue accumulates per-column sum / sum-of-squares for BatchNorm.
// ---------------------------------------------------------------------------
template <int MODE>
__global__ __launch_bounds__(256)
void fc_gemm(const _Float16* __restrict__ Aprev, int ldA,
             const float* __restrict__ prevScale, const float* __restrict__ prevShift,
             const float* __restrict__ lf, const float* __restrict__ sub,
             const float* __restrict__ tgt,
             const float* __restrict__ W, int trueK, int Kpad, int dout, int ldY,
             _Float16* __restrict__ Y,
             float* __restrict__ colSum, float* __restrict__ colSq, int n)
{
    __shared__ _Float16 As[2][64  * LDSS];
    __shared__ _Float16 Bs[2][128 * LDSS];

    const int tid  = threadIdx.x;
    const int lane = tid & 31;
    const int w    = tid >> 5;
    const int wm   = w & 1;        // 2 waves along M
    const int wn   = w >> 1;       // 4 waves along N
    const int m0   = blockIdx.x * 64;
    const int n0   = blockIdx.y * 128;
    const int KT   = Kpad >> 5;

    auto stageA = [&](int kt, int buf) {
        // 64 rows x 32 k = 256 chunks of 8 halves; one chunk per thread
        int row  = tid >> 2;
        int kloc = (tid & 3) * 8;
        int kg   = kt * 32 + kloc;
        int gr   = m0 + row;
        v8h out = {};
        if (gr < n) {
            if (MODE == 1) {
                if (kg + 8 <= trueK) {
                    v8h va = *(const v8h*)(Aprev + (size_t)gr * ldA + kg);
                    const float4* scp = (const float4*)(prevScale + kg);
                    const float4* shp = (const float4*)(prevShift + kg);
                    float4 sc0 = scp[0], sc1 = scp[1];
                    float4 sh0 = shp[0], sh1 = shp[1];
                    float sc[8] = {sc0.x, sc0.y, sc0.z, sc0.w, sc1.x, sc1.y, sc1.z, sc1.w};
                    float sh[8] = {sh0.x, sh0.y, sh0.z, sh0.w, sh1.x, sh1.y, sh1.z, sh1.w};
                    #pragma unroll
                    for (int j = 0; j < 8; ++j)
                        out[j] = (_Float16)lrelu((float)va[j] * sc[j] + sh[j]);
                    if (kt + 1 < KT)   // speculative prefetch of next K-tile
                        __builtin_prefetch(Aprev + (size_t)gr * ldA + kg + 32, 0, 1);
                }
            } else {
                // concat [lf | is_subgoal | is_target | 0-pad], fp32 -> f16
                const float* lfrow = lf + (size_t)gr * 772;
                if (kg + 8 <= 772) {               // bulk: two aligned float4 loads
                    const float4* sp = (const float4*)(lfrow + kg);
                    float4 f0 = sp[0], f1 = sp[1];
                    out[0] = (_Float16)f0.x; out[1] = (_Float16)f0.y;
                    out[2] = (_Float16)f0.z; out[3] = (_Float16)f0.w;
                    out[4] = (_Float16)f1.x; out[5] = (_Float16)f1.y;
                    out[6] = (_Float16)f1.z; out[7] = (_Float16)f1.w;
                } else if (kg < 774) {             // single boundary chunk (kg==768)
                    #pragma unroll
                    for (int j = 0; j < 8; ++j) {
                        int k = kg + j;
                        float f = 0.f;
                        if (k < 772)       f = lfrow[k];
                        else if (k == 772) f = sub[gr];
                        else if (k == 773) f = tgt[gr];
                        out[j] = (_Float16)f;
                    }
                }                                   // kg >= 774: stays zero
            }
        }
        *(v8h*)(&As[buf][row * LDSS + kloc]) = out;
    };

    auto stageB = [&](int kt, int buf) {
        // 32 k x 128 n = 512 chunks of 8 n-values; two chunks per thread.
        // dout and nseg are multiples of 8 => the n-guard is chunk-uniform:
        // each chunk is either a clean float4 x2 load or all zeros.
        #pragma unroll
        for (int i = 0; i < 2; ++i) {
            int c    = tid + i * 256;
            int k    = c >> 4;
            int nseg = (c & 15) * 8;
            int kg   = kt * 32 + k;
            int ng0  = n0 + nseg;
            v8h out = {};
            if (kg < trueK && ng0 + 8 <= dout) {
                const float4* wp = (const float4*)(W + (size_t)kg * dout + ng0);
                float4 w0 = wp[0], w1 = wp[1];
                out[0] = (_Float16)w0.x; out[1] = (_Float16)w0.y;
                out[2] = (_Float16)w0.z; out[3] = (_Float16)w0.w;
                out[4] = (_Float16)w1.x; out[5] = (_Float16)w1.y;
                out[6] = (_Float16)w1.z; out[7] = (_Float16)w1.w;
            }
            // transposed scatter Bs[n][k] so a lane's B fragment is K-contiguous
            #pragma unroll
            for (int j = 0; j < 8; ++j)
                Bs[buf][(nseg + j) * LDSS + k] = out[j];
        }
    };

    v8f acc[2][2] = {};

    stageA(0, 0);
    stageB(0, 0);
    __syncthreads();

    const int kA = (lane & 16) ? 16 : 0;   // A frag: K {0..15} / {16..31}
    const int kB = (lane & 16) ? 16 : 0;   // B frag: K {0..15} / {16..31}

    for (int kt = 0; kt < KT; ++kt) {
        if (kt + 1 < KT) { stageA(kt + 1, (kt + 1) & 1); stageB(kt + 1, (kt + 1) & 1); }
        const int buf = kt & 1;

        v16h a[2], b[2];
        #pragma unroll
        for (int mi = 0; mi < 2; ++mi) {
            int row = wm * 32 + mi * 16 + (lane & 15);
            v8h lo = *(const v8h*)(&As[buf][row * LDSS + kA]);
            v8h hi = *(const v8h*)(&As[buf][row * LDSS + kA + 8]);
            #pragma unroll
            for (int j = 0; j < 8; ++j) { a[mi][j] = lo[j]; a[mi][j + 8] = hi[j]; }
        }
        #pragma unroll
        for (int ni = 0; ni < 2; ++ni) {
            int nr = wn * 32 + ni * 16 + (lane & 15);
            v8h b0 = *(const v8h*)(&Bs[buf][nr * LDSS + kB]);
            v8h b1 = *(const v8h*)(&Bs[buf][nr * LDSS + kB + 8]);
            #pragma unroll
            for (int j = 0; j < 8; ++j) { b[ni][j] = b0[j]; b[ni][j + 8] = b1[j]; }
        }
        #pragma unroll
        for (int mi = 0; mi < 2; ++mi)
            #pragma unroll
            for (int ni = 0; ni < 2; ++ni)
                acc[mi][ni] = __builtin_amdgcn_wmma_f32_16x16x32_f16(
                    false, a[mi], false, b[ni], (short)0, acc[mi][ni], false, false);
        __syncthreads();
    }

    // Epilogue: store pre-BN Y (f16) + accumulate BN column stats.
    // C layout: VGPR v -> row v (lanes 0-15) / v+8 (lanes 16-31), col = lane&15.
    #pragma unroll
    for (int mi = 0; mi < 2; ++mi) {
        int rowBase = m0 + wm * 32 + mi * 16 + ((lane & 16) ? 8 : 0);
        #pragma unroll
        for (int ni = 0; ni < 2; ++ni) {
            int col = n0 + wn * 32 + ni * 16 + (lane & 15);
            if (col < dout) {
                #pragma unroll
                for (int v = 0; v < 8; ++v) {
                    int r = rowBase + v;
                    if (r < n) Y[(size_t)r * ldY + col] = (_Float16)acc[mi][ni][v];
                }
            }
        }
    }
    #pragma unroll
    for (int ni = 0; ni < 2; ++ni) {
        int col = n0 + wn * 32 + ni * 16 + (lane & 15);
        float s = 0.f, q = 0.f;
        #pragma unroll
        for (int mi = 0; mi < 2; ++mi)
            #pragma unroll
            for (int v = 0; v < 8; ++v) { float f = acc[mi][ni][v]; s += f; q += f * f; }
        s += __shfl_xor(s, 16, 32);    // combine lane/lane^16 (same column)
        q += __shfl_xor(q, 16, 32);
        if ((lane < 16) && col < dout) {
            gAtomAdd(&colSum[col], s);
            gAtomAdd(&colSq[col],  q);
        }
    }
}

// ---------------------------------------------------------------------------
// BN finalize: (sum, sumsq) -> (scale, shift).  Slot layout (floats):
//   [0..511]=sum  [512..1023]=sumsq  [1024..1535]=scale  [1536..2047]=shift
// ---------------------------------------------------------------------------
__global__ void bn_finalize(const float* __restrict__ g, const float* __restrict__ be,
                            float* __restrict__ slot, int dout, float invN)
{
    int j = threadIdx.x;
    if (j < dout) {
        float mean = slot[j] * invN;
        float var  = slot[512 + j] * invN - mean * mean;
        var = var < 0.f ? 0.f : var;
        float sc = g[j] * rsqrtf(var + EPSBN);
        slot[1024 + j] = sc;
        slot[1536 + j] = be[j] - mean * sc;
    }
}

// Apply BN+LeakyReLU to FC7 output (f16, ld=8) -> f32 node features h0
__global__ void apply_fc7(const _Float16* __restrict__ Y, const float* __restrict__ slot,
                          float* __restrict__ h, int n)
{
    int idx = blockIdx.x * blockDim.x + threadIdx.x;
    if (idx >= n * 8) return;
    int j = idx & 7;
    float f = (float)Y[idx] * slot[1024 + j] + slot[1536 + j];
    h[idx] = lrelu(f);
}

// Degree of symmetrized graph: each undirected edge hits both endpoints
__global__ void deg_kernel(const int* __restrict__ ed, int E, float* __restrict__ deg)
{
    int t = blockIdx.x * blockDim.x + threadIdx.x;
    if (t >= E) return;
    gAtomAdd(&deg[ed[t]],     1.f);
    gAtomAdd(&deg[ed[E + t]], 1.f);
}

// Scatter: agg[dst] += h[src] over 2E directed edges (8 features)
__global__ void edge_scatter(const int* __restrict__ ed, int E,
                             const float* __restrict__ h, float* __restrict__ agg)
{
    int t = blockIdx.x * blockDim.x + threadIdx.x;
    if (t >= 2 * E) return;
    int s, d;
    if (t < E) { s = ed[t];           d = ed[E + t]; }
    else       { s = ed[E + (t - E)]; d = ed[t - E]; }
    const float4* hp = (const float4*)(h + (size_t)s * 8);
    float4 h0 = hp[0], h1 = hp[1];
    float* ap = agg + (size_t)d * 8;
    gAtomAdd(ap + 0, h0.x); gAtomAdd(ap + 1, h0.y);
    gAtomAdd(ap + 2, h0.z); gAtomAdd(ap + 3, h0.w);
    gAtomAdd(ap + 4, h1.x); gAtomAdd(ap + 5, h1.y);
    gAtomAdd(ap + 6, h1.z); gAtomAdd(ap + 7, h1.w);
}

// Per-node SAGEConv: y = (agg/deg)@Wl + h@Wr + bl, plus LDS-reduced BN stats
__global__ __launch_bounds__(256)
void conv_node(const float* __restrict__ agg, const float* __restrict__ deg,
               const float* __restrict__ h,
               const float* __restrict__ Wl, const float* __restrict__ bl,
               const float* __restrict__ Wr, int dout,
               float* __restrict__ y, float* __restrict__ slot, int n)
{
    __shared__ float sWl[64], sWr[64], sBl[8], sSum[8], sSq[8];
    int tid = threadIdx.x;
    if (tid < 8 * dout) { sWl[tid] = Wl[tid]; sWr[tid] = Wr[tid]; }
    if (tid < dout)     { sBl[tid] = bl[tid]; sSum[tid] = 0.f; sSq[tid] = 0.f; }
    __syncthreads();

    int i = blockIdx.x * 256 + tid;
    if (i < n) {
        float d = deg[i]; d = d < 1.f ? 1.f : d;
        float inv = 1.f / d;
        float a[8], hh[8];
        #pragma unroll
        for (int k = 0; k < 8; ++k) {
            a[k]  = agg[(size_t)i * 8 + k] * inv;
            hh[k] = h[(size_t)i * 8 + k];
        }
        for (int j = 0; j < dout; ++j) {
            float s = sBl[j];
            #pragma unroll
            for (int k = 0; k < 8; ++k) s += a[k] * sWl[k * dout + j] + hh[k] * sWr[k * dout + j];
            y[(size_t)i * 8 + j] = s;
            atomicAdd(&sSum[j], s);        // LDS: native ds_add_f32
            atomicAdd(&sSq[j],  s * s);
        }
    }
    __syncthreads();
    if (tid < dout) {
        gAtomAdd(&slot[tid],       sSum[tid]);
        gAtomAdd(&slot[512 + tid], sSq[tid]);
    }
}

// BN+LeakyReLU for conv output (stride-8 buffers, zero-pad beyond dout)
__global__ void conv_apply(const float* __restrict__ y, const float* __restrict__ slot,
                           int dout, float* __restrict__ hn, int n)
{
    int idx = blockIdx.x * blockDim.x + threadIdx.x;
    if (idx >= n * 8) return;
    int j = idx & 7;
    float v = 0.f;
    if (j < dout) v = lrelu(y[idx] * slot[1024 + j] + slot[1536 + j]);
    hn[idx] = v;
}

__global__ void classifier(const float* __restrict__ h, const float* __restrict__ cw,
                           const float* __restrict__ cb, float* __restrict__ out, int n)
{
    int i = blockIdx.x * blockDim.x + threadIdx.x;
    if (i >= n) return;
    float s = cb[0];
    #pragma unroll
    for (int k = 0; k < 4; ++k) s += h[(size_t)i * 8 + k] * cw[k];
    out[i] = s;
}

// ---------------------------------------------------------------------------
extern "C" void kernel_launch(void* const* d_in, const int* in_sizes, int n_in,
                              void* d_out, int out_size, void* d_ws, size_t ws_size,
                              hipStream_t stream)
{
    (void)n_in; (void)out_size; (void)ws_size;

    const float* lf  = (const float*)d_in[0];
    const float* sub = (const float*)d_in[1];
    const float* tgt = (const float*)d_in[2];
    const int*   ed  = (const int*)d_in[3];
    const int N = in_sizes[1];
    const int E = in_sizes[3] / 2;

    // fc i: W=4+4i b=5+4i g=6+4i be=7+4i ; conv l: Wl=32+5l bl=33 Wr=34 g=35 be=36
    const float* clsW = (const float*)d_in[52];
    const float* clsB = (const float*)d_in[53];

    // ---- workspace layout -------------------------------------------------
    char* ws = (char*)d_ws;
    _Float16* YA   = (_Float16*)ws;                         // N*512 f16
    _Float16* YB   = (_Float16*)(ws + (size_t)N * 512 * 2); // N*256 f16
    char* p        = ws + (size_t)N * 512 * 2 + (size_t)N * 256 * 2;
    float* stats   = (float*)p;           p += 11 * 2048 * sizeof(float);  // 11 BN slots
    float* deg     = (float*)p;           p += (size_t)N * sizeof(float);
    float* agg     = (float*)p;           p += (size_t)N * 8 * sizeof(float);
    float* h0      = (float*)p;           p += (size_t)N * 8 * sizeof(float);
    float* h1      = (float*)p;           p += (size_t)N * 8 * sizeof(float);
    float* yconv   = (float*)p;

    const int T = 256;
    // zero all BN stat accumulators + degree (fresh every launch / replay)
    zero_f32<<<(11 * 2048 + T - 1) / T, T, 0, stream>>>(stats, 11 * 2048);
    zero_f32<<<(N + T - 1) / T, T, 0, stream>>>(deg, N);
    deg_kernel<<<(E + T - 1) / T, T, 0, stream>>>(ed, E, deg);

    // ---- FC stack ---------------------------------------------------------
    const int dims[8] = {774, 512, 256, 128, 64, 32, 16, 8};
    const _Float16* Ain = nullptr;
    int ldA = 0;
    _Float16* Ybufs[2] = {YA, YB};
    for (int i = 0; i < 7; ++i) {
        int dout  = dims[i + 1];
        int trueK = (i == 0) ? 774 : dims[i];
        int Kpad  = (i == 0) ? 800 : (dims[i] < 32 ? 32 : dims[i]);
        _Float16* Yout = Ybufs[i & 1];
        float* slot = stats + (size_t)i * 2048;
        const float* pSc = (i == 0) ? nullptr : stats + (size_t)(i - 1) * 2048 + 1024;
        const float* pSh = (i == 0) ? nullptr : stats + (size_t)(i - 1) * 2048 + 1536;
        dim3 grid((N + 63) / 64, (dout + 127) / 128);
        const float* Wi = (const float*)d_in[4 + 4 * i];
        if (i == 0)
            fc_gemm<0><<<grid, 256, 0, stream>>>(Ain, ldA, pSc, pSh, lf, sub, tgt,
                                                 Wi, trueK, Kpad, dout, dout, Yout,
                                                 slot, slot + 512, N);
        else
            fc_gemm<1><<<grid, 256, 0, stream>>>(Ain, ldA, pSc, pSh, lf, sub, tgt,
                                                 Wi, trueK, Kpad, dout, dout, Yout,
                                                 slot, slot + 512, N);
        bn_finalize<<<1, 512, 0, stream>>>((const float*)d_in[6 + 4 * i],
                                           (const float*)d_in[7 + 4 * i],
                                           slot, dout, 1.0f / (float)N);
        Ain = Yout; ldA = dout;
    }
    apply_fc7<<<(N * 8 + T - 1) / T, T, 0, stream>>>(Ain, stats + (size_t)6 * 2048, h0, N);

    // ---- SAGEConv stack ---------------------------------------------------
    const int cdout[4] = {8, 8, 8, 4};
    float* hc = h0;
    float* hn = h1;
    for (int l = 0; l < 4; ++l) {
        float* slot = stats + (size_t)(7 + l) * 2048;
        zero_f32<<<(N * 8 + T - 1) / T, T, 0, stream>>>(agg, N * 8);
        edge_scatter<<<(2 * E + T - 1) / T, T, 0, stream>>>(ed, E, hc, agg);
        conv_node<<<(N + T - 1) / T, T, 0, stream>>>(agg, deg, hc,
            (const float*)d_in[32 + 5 * l], (const float*)d_in[33 + 5 * l],
            (const float*)d_in[34 + 5 * l], cdout[l], yconv, slot, N);
        bn_finalize<<<1, 512, 0, stream>>>((const float*)d_in[35 + 5 * l],
                                           (const float*)d_in[36 + 5 * l],
                                           slot, cdout[l], 1.0f / (float)N);
        conv_apply<<<(N * 8 + T - 1) / T, T, 0, stream>>>(yconv, slot, cdout[l], hn, N);
        float* t = hc; hc = hn; hn = t;
    }

    classifier<<<(N + T - 1) / T, T, 0, stream>>>(hc, clsW, clsB, (float*)d_out, N);
}
